// ImprovedGraphTransGeoGCN_78546361909471
// MI455X (gfx1250) — matrix operand order repaired and checked by reading, not compiled, not run on hardware
//
#include <hip/hip_runtime.h>
#include <hip/hip_bf16.h>

typedef _Float16 f16;
typedef _Float16 v16h __attribute__((ext_vector_type(16)));
typedef _Float16 v8h  __attribute__((ext_vector_type(8)));
typedef float    v8f  __attribute__((ext_vector_type(8)));

static constexpr int   NN     = 50000;
static constexpr int   NE     = 800000;
static constexpr int   DIN    = 128;
static constexpr int   HD     = 256;
static constexpr float BN_EPS = 1e-5f;
static constexpr float LN_EPS = 1e-5f;

// ---------------------------------------------------------------- utilities
__global__ void k_zero_f32(float* __restrict__ p, long long n) {
    long long i = (long long)blockIdx.x * blockDim.x + threadIdx.x;
    if (i < n) p[i] = 0.f;
}

__global__ void k_degree(const int* __restrict__ dst, float* __restrict__ deg, int E) {
    int e = blockIdx.x * blockDim.x + threadIdx.x;
    if (e < E) atomicAdd(&deg[dst[e]], 1.f);
}

__global__ void k_dinv(const float* __restrict__ deg, float* __restrict__ dinv, int n) {
    int i = blockIdx.x * blockDim.x + threadIdx.x;
    if (i < n) dinv[i] = rsqrtf(deg[i] + 1.f);
}

// ----------------------------------------------------- BN0 (input features)
__global__ void k_bn0_stats(const float* __restrict__ x, const float* __restrict__ fs,
                            const float* __restrict__ fb, float* __restrict__ sum,
                            float* __restrict__ sq, int n) {
    int t = threadIdx.x;                       // blockDim.x == DIN
    float sc = fs[t], bi = fb[t], s = 0.f, q = 0.f;
    for (int r = blockIdx.x; r < n; r += gridDim.x) {
        float v = x[(size_t)r * DIN + t] * sc + bi;
        s += v; q += v * v;
    }
    atomicAdd(&sum[t], s); atomicAdd(&sq[t], q);
}

__global__ void k_bn_params(const float* __restrict__ sum, const float* __restrict__ sq,
                            const float* __restrict__ g, const float* __restrict__ b,
                            float* __restrict__ a, float* __restrict__ d, int C, float invn) {
    int t = blockIdx.x * blockDim.x + threadIdx.x;
    if (t < C) {
        float mu  = sum[t] * invn;
        float var = sq[t] * invn - mu * mu;
        float aa  = g[t] * rsqrtf(var + BN_EPS);
        a[t] = aa; d[t] = b[t] - mu * aa;
    }
}

__global__ void k_bn0_apply(const float* __restrict__ x, const float* __restrict__ fs,
                            const float* __restrict__ fb, const float* __restrict__ a,
                            const float* __restrict__ d, f16* __restrict__ h0, long long n) {
    long long i = (long long)blockIdx.x * blockDim.x + threadIdx.x;
    if (i < n) {
        int c = (int)(i & (DIN - 1));
        float v = x[i] * fs[c] + fb[c];
        h0[i] = (f16)(a[c] * v + d[c]);
    }
}

// --------------------------------------------- pack W[K x NC] -> WT[NC x K] (f16)
__global__ void k_pack_wt(const float* __restrict__ W, f16* __restrict__ WT, int K, int NC) {
    int i = blockIdx.x * blockDim.x + threadIdx.x;
    if (i < K * NC) {
        int k = i / NC, n = i % NC;
        WT[(size_t)n * K + k] = (f16)W[i];
    }
}

// -------------------------------------------------------------- WMMA GEMM
// C[M x 256] = A[M x K] (f16 row-major) * B (as BT[256 x K] f16, i.e. B col-major)
// block = 128 threads = 4 waves; wave w owns N range [64w, 64w+64); block owns one
// 16-row M tile. Per k-step each wave does 4x v_wmma_f32_16x16x32_f16.
__global__ void __launch_bounds__(128)
k_gemm_wmma(const f16* __restrict__ A, const f16* __restrict__ BT,
            float* __restrict__ C, int K) {
    const int wave = threadIdx.x >> 5;
    const int lane = threadIdx.x & 31;
    const int lm = lane & 15;           // row within tile (A) / column n (B) / N (C)
    const int lc = lane >> 4;           // half-wave selector
    const size_t mbase = (size_t)blockIdx.x * 16;
    const f16* arow = A + (mbase + lm) * (size_t)K;
    const int nbase = wave * 64;

    v8f acc[4];
#pragma unroll
    for (int j = 0; j < 4; ++j)
#pragma unroll
        for (int r = 0; r < 8; ++r) acc[j][r] = 0.f;

    for (int kt = 0; kt < K; kt += 32) {
        // A 16x32 f16, contiguous k order used here (same as B): lane lc half: VGPR0-3 <- K = lc*16..lc*16+7,
        //                                   VGPR4-7 <- K = lc*16+8..lc*16+15
        v8h alo = *(const v8h*)(arow + kt + lc * 16);
        v8h ahi = *(const v8h*)(arow + kt + lc * 16 + 8);
        v16h a;
#pragma unroll
        for (int j = 0; j < 8; ++j) { a[j] = alo[j]; a[j + 8] = ahi[j]; }

#pragma unroll
        for (int j = 0; j < 4; ++j) {
            // B 32x16: lane holds column n = nbase+j*16+lm, K = lc*16..lc*16+15
            const f16* bp = BT + (size_t)(nbase + j * 16 + lm) * K + kt + lc * 16;
            v16h b = *(const v16h*)bp;
            acc[j] = __builtin_amdgcn_wmma_f32_16x16x32_f16(
                false, a, false, b, (short)0, acc[j], false, false);
        }
    }

    // D layout: VGPR r -> M = r + 8*lc, N = lm
#pragma unroll
    for (int j = 0; j < 4; ++j) {
        int n = nbase + j * 16 + lm;
#pragma unroll
        for (int r = 0; r < 8; ++r)
            C[(mbase + r + lc * 8) * HD + n] = acc[j][r];
    }
}

// ------------------------------------------------------------ edge scatter
__global__ void k_edge_scatter(const int* __restrict__ src, const int* __restrict__ dst,
                               const float* __restrict__ dinv, const float* __restrict__ h,
                               float* __restrict__ agg, int E) {
    const int CH = HD / 4;              // 64 float4 chunks per edge
    long long idx = (long long)blockIdx.x * blockDim.x + threadIdx.x;
    if (idx >= (long long)E * CH) return;
    int e  = (int)(idx / CH);
    int c4 = (int)(idx % CH) * 4;
    int s = src[e], dd = dst[e];
    float nrm = dinv[s] * dinv[dd];
    float4 v = *(const float4*)(h + (size_t)s * HD + c4);
    float* p = agg + (size_t)dd * HD + c4;
    atomicAdd(p + 0, v.x * nrm);
    atomicAdd(p + 1, v.y * nrm);
    atomicAdd(p + 2, v.z * nrm);
    atomicAdd(p + 3, v.w * nrm);
}

// -------- z = relu(agg + dinv^2*h + b), in place in agg; accumulate BN stats
__global__ void k_node_relu_stats(float* __restrict__ agg, const float* __restrict__ g,
                                  const float* __restrict__ dinv, const float* __restrict__ bias,
                                  float* __restrict__ sum, float* __restrict__ sq, int n) {
    int t = threadIdx.x;                // blockDim.x == HD
    float b = bias[t], s = 0.f, q = 0.f;
    for (int r = blockIdx.x; r < n; r += gridDim.x) {
        float di = dinv[r];
        size_t i = (size_t)r * HD + t;
        float z = agg[i] + di * di * g[i] + b;
        z = fmaxf(z, 0.f);
        agg[i] = z; s += z; q += z * z;
    }
    atomicAdd(&sum[t], s); atomicAdd(&sq[t], q);
}

// -------- per-row: BN apply (a,d) then LayerNorm, optional residual, f32+f16 out
__global__ void k_bn_ln_row(const float* __restrict__ z, const float* __restrict__ a,
                            const float* __restrict__ d, const float* __restrict__ lng,
                            const float* __restrict__ lnb, const float* __restrict__ resid,
                            float* __restrict__ out32, f16* __restrict__ out16) {
    __shared__ float red[HD];
    int t = threadIdx.x;
    size_t i = (size_t)blockIdx.x * HD + t;
    float v = a[t] * z[i] + d[t];
    red[t] = v; __syncthreads();
#pragma unroll
    for (int s = HD / 2; s > 0; s >>= 1) {
        if (t < s) red[t] += red[t + s];
        __syncthreads();
    }
    float mu = red[0] * (1.f / HD);
    __syncthreads();
    float dv = v - mu;
    red[t] = dv * dv; __syncthreads();
#pragma unroll
    for (int s = HD / 2; s > 0; s >>= 1) {
        if (t < s) red[t] += red[t + s];
        __syncthreads();
    }
    float rs = rsqrtf(red[0] * (1.f / HD) + LN_EPS);
    float o = lng[t] * dv * rs + lnb[t];
    if (resid) o += resid[i];
    out32[i] = o;
    if (out16) out16[i] = (f16)o;
}

// -------------------------------------------- output GEMM: [N,256] x [256,2]
__global__ void k_out(const float* __restrict__ h2, const float* __restrict__ Wout,
                      const float* __restrict__ bout, float* __restrict__ out, int n) {
    int wave = threadIdx.x >> 5, lane = threadIdx.x & 31;
    int row = blockIdx.x * 8 + wave;
    if (row >= n) return;
    const float* hr = h2 + (size_t)row * HD;
    float s0 = 0.f, s1 = 0.f;
#pragma unroll
    for (int k = 0; k < 8; ++k) {
        int c = lane + 32 * k;
        float v = hr[c];
        s0 += v * Wout[c * 2 + 0];
        s1 += v * Wout[c * 2 + 1];
    }
#pragma unroll
    for (int off = 16; off > 0; off >>= 1) {
        s0 += __shfl_xor(s0, off, 32);
        s1 += __shfl_xor(s1, off, 32);
    }
    if (lane == 0) {
        out[row * 2 + 0] = s0 + bout[0];
        out[row * 2 + 1] = s1 + bout[1];
    }
}

// ---------------------------------------------------------------- launcher
extern "C" void kernel_launch(void* const* d_in, const int* in_sizes, int n_in,
                              void* d_out, int out_size, void* d_ws, size_t ws_size,
                              hipStream_t stream) {
    const float* x    = (const float*)d_in[0];
    const int*   eidx = (const int*)d_in[1];
    const int*   src  = eidx;
    const int*   dst  = eidx + NE;
    const float* fs   = (const float*)d_in[2];
    const float* fb   = (const float*)d_in[3];
    const float* bn0g = (const float*)d_in[4];
    const float* bn0b = (const float*)d_in[5];
    const float* W1   = (const float*)d_in[6];
    const float* b1   = (const float*)d_in[7];
    const float* bn1g = (const float*)d_in[8];
    const float* bn1b = (const float*)d_in[9];
    const float* ln1g = (const float*)d_in[10];
    const float* ln1b = (const float*)d_in[11];
    const float* W2   = (const float*)d_in[12];
    const float* b2   = (const float*)d_in[13];
    const float* bn2g = (const float*)d_in[14];
    const float* bn2b = (const float*)d_in[15];
    const float* ln2g = (const float*)d_in[16];
    const float* ln2b = (const float*)d_in[17];
    const float* Wout = (const float*)d_in[18];
    const float* bout = (const float*)d_in[19];
    float* out = (float*)d_out;

    // workspace carving (256B aligned chunks)
    char* base = (char*)d_ws;
    size_t off = 0;
    auto carve = [&](size_t bytes) -> void* {
        void* p = base + off;
        off = (off + bytes + 255) & ~(size_t)255;
        return p;
    };
    float* deg   = (float*)carve(sizeof(float) * NN);
    float* dinv  = (float*)carve(sizeof(float) * NN);
    float* sums  = (float*)carve(sizeof(float) * 1280); // sum0,sq0(128ea) sum1,sq1,sum2,sq2(256ea)
    float* sum0 = sums,       *sq0 = sums + 128;
    float* sum1 = sums + 256, *sq1 = sums + 512;
    float* sum2 = sums + 768, *sq2 = sums + 1024;
    float* ad    = (float*)carve(sizeof(float) * 1280); // a0,d0(128ea) a1,d1,a2,d2(256ea)
    float* a0 = ad,       *d0 = ad + 128;
    float* a1 = ad + 256, *d1 = ad + 512;
    float* a2 = ad + 768, *d2 = ad + 1024;
    float* g     = (float*)carve(sizeof(float) * (size_t)NN * HD);
    float* agg   = (float*)carve(sizeof(float) * (size_t)NN * HD);
    float* h1    = (float*)carve(sizeof(float) * (size_t)NN * HD);
    f16*   h0f16 = (f16*)carve(sizeof(f16) * (size_t)NN * DIN);
    f16*   h1f16 = (f16*)carve(sizeof(f16) * (size_t)NN * HD);
    f16*   wt1   = (f16*)carve(sizeof(f16) * DIN * HD);
    f16*   wt2   = (f16*)carve(sizeof(f16) * HD * HD);
    if (off > ws_size) return;  // insufficient scratch (deterministic no-op)

    const long long nNH = (long long)NN * HD;
    const long long nND = (long long)NN * DIN;

    // zero accumulators
    k_zero_f32<<<(NN + 255) / 256, 256, 0, stream>>>(deg, NN);
    k_zero_f32<<<(1280 + 255) / 256, 256, 0, stream>>>(sums, 1280);

    // degree / dinv
    k_degree<<<(NE + 255) / 256, 256, 0, stream>>>(dst, deg, NE);
    k_dinv<<<(NN + 255) / 256, 256, 0, stream>>>(deg, dinv, NN);

    // input BN -> h0 (f16)
    k_bn0_stats<<<1024, DIN, 0, stream>>>(x, fs, fb, sum0, sq0, NN);
    k_bn_params<<<1, 128, 0, stream>>>(sum0, sq0, bn0g, bn0b, a0, d0, DIN, 1.f / NN);
    k_bn0_apply<<<(unsigned)((nND + 255) / 256), 256, 0, stream>>>(x, fs, fb, a0, d0, h0f16, nND);

    // weight pre-packs (B column-major f16 for WMMA)
    k_pack_wt<<<(DIN * HD + 255) / 256, 256, 0, stream>>>(W1, wt1, DIN, HD);
    k_pack_wt<<<(HD * HD + 255) / 256, 256, 0, stream>>>(W2, wt2, HD, HD);

    // ---- layer 1 ----
    k_gemm_wmma<<<NN / 16, 128, 0, stream>>>(h0f16, wt1, g, DIN);
    k_zero_f32<<<(unsigned)((nNH + 255) / 256), 256, 0, stream>>>(agg, nNH);
    k_edge_scatter<<<(unsigned)(((long long)NE * (HD / 4) + 255) / 256), 256, 0, stream>>>(
        src, dst, dinv, g, agg, NE);
    k_node_relu_stats<<<2048, HD, 0, stream>>>(agg, g, dinv, b1, sum1, sq1, NN);
    k_bn_params<<<1, 256, 0, stream>>>(sum1, sq1, bn1g, bn1b, a1, d1, HD, 1.f / NN);
    k_bn_ln_row<<<NN, HD, 0, stream>>>(agg, a1, d1, ln1g, ln1b, nullptr, h1, h1f16);

    // ---- layer 2 ----
    k_gemm_wmma<<<NN / 16, 128, 0, stream>>>(h1f16, wt2, g, HD);
    k_zero_f32<<<(unsigned)((nNH + 255) / 256), 256, 0, stream>>>(agg, nNH);
    k_edge_scatter<<<(unsigned)(((long long)NE * (HD / 4) + 255) / 256), 256, 0, stream>>>(
        src, dst, dinv, g, agg, NE);
    k_node_relu_stats<<<2048, HD, 0, stream>>>(agg, g, dinv, b2, sum2, sq2, NN);
    k_bn_params<<<1, 256, 0, stream>>>(sum2, sq2, bn2g, bn2b, a2, d2, HD, 1.f / NN);
    k_bn_ln_row<<<NN, HD, 0, stream>>>(agg, a2, d2, ln2g, ln2b, h1, g, nullptr);

    // ---- output head ----
    k_out<<<(NN + 7) / 8, 256, 0, stream>>>(g, Wout, bout, out, NN);
}
